// LDConv_5136780886646
// MI455X (gfx1250) — compile-verified, matrix operand order checked by reading, not run on hardware
//
#include <hip/hip_runtime.h>
#include <hip/hip_bf16.h>

// ---------------------------------------------------------------------------
// LDConv forward for MI455X (gfx1250).
// B=4, C=128, H=W=128, N=9, outc=256, stride=1  (hardcoded; matches setup_inputs)
//
// Stage 1: offset = conv3x3(x, w_p) + b_p                  (4,18,128,128) f32
// Stage 2: Xo[c*9+n, (b,y,x)] = bilinear-sample, bf16, pre-swizzled into the
//          WMMA 16x16x32 B-operand lane layout (151 MB -> lives in 192MB L2)
// Stage 3: out = W(256x1152) x Xo(1152x65536) via v_wmma_f32_16x16x32_bf16
// ---------------------------------------------------------------------------

typedef __attribute__((ext_vector_type(16))) __bf16 v16bf;
typedef __attribute__((ext_vector_type(8)))  float  v8f;

#define Bn   4
#define Cc   128
#define Hh   128
#define Ww   128
#define HW   16384      // 128*128
#define NP   9
#define OUTC 256
#define KK   1152       // Cc*NP
#define NN   65536      // Bn*H*W
#define KT   36         // KK/32
#define MT   16         // OUTC/16
#define NT   4096       // NN/16

// workspace layout (bytes)
#define OFF_OFFSET 0u
#define SZ_OFFSET  (4u*18u*16384u*4u)            // 4,718,592
#define OFF_WSW    SZ_OFFSET                     // 4,718,592
#define SZ_WSW     (MT*KT*512u*2u)               // 589,824
#define OFF_XSW    (OFF_WSW + SZ_WSW)            // 5,308,416 (32B aligned)

__device__ __forceinline__ unsigned short f2bf(float f) {
    unsigned u = __builtin_bit_cast(unsigned, f);
    unsigned r = u + 0x7FFFu + ((u >> 16) & 1u);   // round-to-nearest-even
    return (unsigned short)(r >> 16);
}

// ---------------------------------------------------------------------------
// Stage 1: 3x3 offset conv, 128 -> 18 channels, pad 1. One thread = (b,y,x).
// w_p staged through LDS in 16-channel slices (broadcast reads, no conflicts).
// ---------------------------------------------------------------------------
__global__ __launch_bounds__(256) void offset_conv_kernel(
    const float* __restrict__ x, const float* __restrict__ wp,
    const float* __restrict__ bp, float* __restrict__ off) {
    __shared__ float wps[18 * 16 * 9];
    const int tid = blockIdx.x * 256 + threadIdx.x;     // 0..65535
    const int b  = tid >> 14;
    const int y  = (tid >> 7) & 127;
    const int xx = tid & 127;
    float acc[18];
#pragma unroll
    for (int o = 0; o < 18; ++o) acc[o] = 0.f;
    const float* xb = x + (size_t)b * Cc * HW;
    for (int c0 = 0; c0 < Cc; c0 += 16) {
        __syncthreads();
        for (int i = threadIdx.x; i < 18 * 16 * 9; i += 256) {
            int oc = i / 144, rem = i % 144, cc = rem / 9, k = rem % 9;
            wps[i] = wp[(size_t)oc * KK + (size_t)(c0 + cc) * 9 + k];
        }
        __syncthreads();
        for (int cc = 0; cc < 16; ++cc) {
            const float* xc = xb + (size_t)(c0 + cc) * HW;
#pragma unroll
            for (int kh = 0; kh < 3; ++kh) {
                int yy = y + kh - 1;
                if (yy < 0 || yy > 127) continue;
#pragma unroll
                for (int kw = 0; kw < 3; ++kw) {
                    int xw = xx + kw - 1;
                    if (xw < 0 || xw > 127) continue;
                    float v = xc[yy * 128 + xw];
                    int k = kh * 3 + kw;
#pragma unroll
                    for (int o = 0; o < 18; ++o)
                        acc[o] += v * wps[o * 144 + cc * 9 + k];
                }
            }
        }
    }
#pragma unroll
    for (int o = 0; o < 18; ++o)
        off[((size_t)b * 18 + o) * HW + y * 128 + xx] = acc[o] + bp[o];
}

// ---------------------------------------------------------------------------
// Stage 2a: pre-swizzle w_conv into bf16 WMMA A-operand layout.
// A 16x32 bf16 per ISA: lane (l<16 -> Kbase 0, l>=16 -> Kbase 8), M = l%16,
// element j: K = Kbase + j + (j>=8 ? 8 : 0). One block = contiguous 1KB tile.
// ---------------------------------------------------------------------------
__global__ __launch_bounds__(256) void wswizzle_kernel(
    const float* __restrict__ wc, unsigned short* __restrict__ Wsw) {
    int t = blockIdx.x * 256 + threadIdx.x;            // 0..294911
    int j    = t & 15;
    int lane = (t >> 4) & 31;
    int blk  = t >> 9;                                  // m_tile*KT + k_tile
    int kt = blk % KT;
    int mt = blk / KT;
    int M  = mt * 16 + (lane & 15);
    int Kl = ((lane >> 4) << 3) + j + ((j >= 8) ? 8 : 0);
    int K  = kt * 32 + Kl;                              // K = c*9 + n
    Wsw[t] = f2bf(wc[(size_t)M * KK + K]);              // w_conv (256,128,9,1) flat
}

// ---------------------------------------------------------------------------
// Stage 2b: bilinear deformable sampling -> bf16, scattered into the WMMA
// B-operand layout (mirror of A with rows<->cols).  One thread = (Nidx, n),
// loops over all 128 channels. Lanes are consecutive x -> coalesced gathers.
// ---------------------------------------------------------------------------
__global__ __launch_bounds__(256) void sample_kernel(
    const float* __restrict__ x, const float* __restrict__ off,
    unsigned short* __restrict__ Xsw) {
    const int Nidx = blockIdx.x * 256 + threadIdx.x;    // 0..65535
    const int n    = blockIdx.y;                        // 0..8
    const int b  = Nidx >> 14;
    const int y  = (Nidx >> 7) & 127;
    const int xx = Nidx & 127;

    float offx = off[((size_t)b * 18 + n)      * HW + y * 128 + xx];
    float offy = off[((size_t)b * 18 + 9 + n)  * HW + y * 128 + xx];
    float px = (float)y  + (float)(n / 3) + offx;       // p = p0 + p_n + offset
    float py = (float)xx + (float)(n % 3) + offy;
    float fx = floorf(px), fy = floorf(py);
    int h0 = min(max((int)fx,     0), Hh - 1);
    int h1 = min(max((int)fx + 1, 0), Hh - 1);
    int w0 = min(max((int)fy,     0), Ww - 1);
    int w1 = min(max((int)fy + 1, 0), Ww - 1);
    float pxc = fminf(fmaxf(px, 0.f), (float)(Hh - 1));
    float pyc = fminf(fmaxf(py, 0.f), (float)(Ww - 1));
    float gx0 = 1.f + ((float)h0 - pxc);
    float gx1 = 1.f - ((float)h1 - pxc);
    float gy0 = 1.f + ((float)w0 - pyc);
    float gy1 = 1.f - ((float)w1 - pyc);
    float g_lt = gx0 * gy0, g_rb = gx1 * gy1, g_lb = gx0 * gy1, g_rt = gx1 * gy0;

    const float* xb = x + (size_t)b * Cc * HW;
    const int i00 = h0 * 128 + w0, i11 = h1 * 128 + w1;
    const int i01 = h0 * 128 + w1, i10 = h1 * 128 + w0;
    const int n_tile = Nidx >> 4, nn = Nidx & 15;

    for (int c = 0; c < Cc; ++c) {
        const float* xc = xb + (size_t)c * HW;
        float v = g_lt * xc[i00] + g_rb * xc[i11] + g_lb * xc[i01] + g_rt * xc[i10];
        int K  = c * 9 + n;
        int kt = K >> 5, kk = K & 31;
        int lane_b = nn + (((kk >> 3) & 1) << 4);        // inverse of lane layout
        int j      = (kk & 7) + ((kk >> 4) << 3);
        Xsw[((size_t)n_tile * KT + kt) * 512 + lane_b * 16 + j] = f2bf(v);
    }
}

// ---------------------------------------------------------------------------
// Stage 3: GEMM D(256 x 65536) = W(256 x 1152) x Xo(1152 x 65536), bf16->f32.
// 8 waves/block as 2(M) x 4(N); each wave: 2 A tiles x 4 B tiles -> 8 WMMA
// per K step, 36 K steps.  Operands land directly in v16bf registers via
// coalesced 32B/lane loads (pre-swizzled), B tiles stream from L2.
// ---------------------------------------------------------------------------
__global__ __launch_bounds__(256) void gemm_kernel(
    const unsigned short* __restrict__ Wsw,
    const unsigned short* __restrict__ Xsw,
    float* __restrict__ out) {
    const int lane = threadIdx.x & 31;
    const int wave = threadIdx.x >> 5;
    const int wm = wave >> 2;                     // 0..1
    const int wn = wave & 3;                      // 0..3
    const int mt0 = blockIdx.y * 4 + wm * 2;      // 2 M-tiles: mt0, mt0+1
    const int nt0 = blockIdx.x * 16 + wn * 4;     // 4 N-tiles

    v8f acc[2][4];
#pragma unroll
    for (int i = 0; i < 2; ++i)
#pragma unroll
        for (int t = 0; t < 4; ++t)
            acc[i][t] = (v8f){0.f, 0.f, 0.f, 0.f, 0.f, 0.f, 0.f, 0.f};

    for (int kt = 0; kt < KT; ++kt) {
        v16bf a0 = *(const v16bf*)(Wsw + (((size_t)mt0 * KT + kt) * 512 + lane * 16));
        v16bf a1 = *(const v16bf*)(Wsw + (((size_t)(mt0 + 1) * KT + kt) * 512 + lane * 16));
#pragma unroll
        for (int t = 0; t < 4; ++t) {
            v16bf bv = *(const v16bf*)(Xsw + (((size_t)(nt0 + t) * KT + kt) * 512 + lane * 16));
            acc[0][t] = __builtin_amdgcn_wmma_f32_16x16x32_bf16(
                false, a0, false, bv, (short)0, acc[0][t], false, false);
            acc[1][t] = __builtin_amdgcn_wmma_f32_16x16x32_bf16(
                false, a1, false, bv, (short)0, acc[1][t], false, false);
        }
    }

    // C/D layout: VGPR r, lane l -> M = r + 8*(l/16), N = l%16
    const int col    = lane & 15;
    const int rowoff = (lane >> 4) << 3;
#pragma unroll
    for (int i = 0; i < 2; ++i) {
        int o0 = (mt0 + i) * 16 + rowoff;
#pragma unroll
        for (int t = 0; t < 4; ++t) {
            int Nidx = (nt0 + t) * 16 + col;
            int bb = Nidx >> 14;
            int y  = (Nidx >> 7) & 127;
            int xx = Nidx & 127;
            float* op = out + (size_t)bb * OUTC * HW + y * 128 + xx;
#pragma unroll
            for (int r = 0; r < 8; ++r)
                op[(size_t)(o0 + r) * HW] = acc[i][t][r];
        }
    }
}

// ---------------------------------------------------------------------------
extern "C" void kernel_launch(void* const* d_in, const int* in_sizes, int n_in,
                              void* d_out, int out_size, void* d_ws, size_t ws_size,
                              hipStream_t stream) {
    (void)in_sizes; (void)n_in; (void)out_size; (void)ws_size;
    const float* x      = (const float*)d_in[0];   // (4,128,128,128)
    const float* w_p    = (const float*)d_in[1];   // (18,128,3,3)
    const float* b_p    = (const float*)d_in[2];   // (18,)
    const float* w_conv = (const float*)d_in[3];   // (256,128,9,1)
    // d_in[4] = num_param (9), d_in[5] = stride (1) -- hardcoded above
    float* out = (float*)d_out;                    // (4,256,128,128)

    char* ws = (char*)d_ws;
    float*          offbuf = (float*)(ws + OFF_OFFSET);
    unsigned short* Wsw    = (unsigned short*)(ws + OFF_WSW);
    unsigned short* Xsw    = (unsigned short*)(ws + OFF_XSW);

    offset_conv_kernel<<<dim3(256), dim3(256), 0, stream>>>(x, w_p, b_p, offbuf);
    wswizzle_kernel<<<dim3((MT * KT * 512) / 256), dim3(256), 0, stream>>>(w_conv, Wsw);
    sample_kernel<<<dim3(NN / 256, NP), dim3(256), 0, stream>>>(x, offbuf, Xsw);
    gemm_kernel<<<dim3(NT / 16, MT / 4), dim3(256), 0, stream>>>(Wsw, Xsw, out);
}
